// SelfAttention_57191784513648
// MI455X (gfx1250) — compile-verified, matrix operand order checked by reading, not run on hardware
//
#include <hip/hip_runtime.h>
#include <hip/hip_bf16.h>

// Self-attention, MI455X (gfx1250, wave32, WMMA).
// proj_qkv (3x GEMM, f32->bf16 on the fly) -> attn_scores (QK^T*scale+mask,
// double-buffered async-to-LDS bf16 WMMA) -> softmax_rows (in-place P bf16)
// -> attn_out (P@V, same async-LDS WMMA pipeline) -> f32 output.
// Workspace: Qb,Kb,Vt bf16 (3*32MB) + S f32 (128MB) ~= 235MB.

typedef __attribute__((ext_vector_type(16))) __bf16 v16bf;
typedef __attribute__((ext_vector_type(8)))  __bf16 v8bf;
typedef __attribute__((ext_vector_type(8)))  float  v8f;

static constexpr int BATCH = 8;
static constexpr int SEQ   = 2048;
static constexpr int DIM   = 1024;

__device__ __forceinline__ v8f wmma_bf16(v16bf a, v16bf b, v8f c) {
  // 8 args: (neg_a, A, neg_b, B, c_mod, C, reuse_a, reuse_b)
  return __builtin_amdgcn_wmma_f32_16x16x32_bf16(false, a, false, b, (short)0, c,
                                                 false, false);
}

// A/B fragment (16x32 / 32x16 bf16): lanes 0-15 hold K {0..7,16..23}, lanes
// 16-31 hold K {8..15,24..31}; caller pre-offsets by (lane>>4)*8 so we always
// read p[0..7] and p[16..23].
__device__ __forceinline__ v16bf frag_from_f32(const float* __restrict__ p) {
  v16bf f;
#pragma unroll
  for (int i = 0; i < 8; ++i) f[i] = (__bf16)p[i];
#pragma unroll
  for (int i = 0; i < 8; ++i) f[i + 8] = (__bf16)p[i + 16];
  return f;
}

__device__ __forceinline__ v16bf frag_from_bf16(const __bf16* p) {
  v8bf lo = *(const v8bf*)p;
  v8bf hi = *(const v8bf*)(p + 16);
  v16bf f;
#pragma unroll
  for (int i = 0; i < 8; ++i) { f[i] = lo[i]; f[i + 8] = hi[i]; }
  return f;
}

// LDS aperture: generic shared address low 32 bits == LDS byte offset.
__device__ __forceinline__ uint32_t lds_off_u32(const void* p) {
  return (uint32_t)(uintptr_t)p;
}

// GLOBAL_LOAD_ASYNC_TO_LDS_B128, GVS mode: saddr(64) + vaddr(32-bit byte off).
// Tracked by ASYNCcnt; 16B per lane -> 512B per wave per issue.
__device__ __forceinline__ void async_b128(uint32_t lds, const void* sbase,
                                           uint32_t byteoff) {
  asm volatile("global_load_async_to_lds_b128 %0, %1, %2"
               :: "v"(lds), "v"(byteoff), "s"(sbase) : "memory");
}

// ---------------------------------------------------------------------------
// Kernel 1: QKV projections. y = x @ W^T + b. M=B*L, N=K=D. grid.z selects W.
// z==2 (V) stored transposed per batch: Vt[b][d][l] (K-contiguous for PV).
// ---------------------------------------------------------------------------
__global__ __launch_bounds__(256) void proj_qkv(
    const float* __restrict__ x,
    const float* __restrict__ Wq, const float* __restrict__ bq,
    const float* __restrict__ Wk, const float* __restrict__ bk,
    const float* __restrict__ Wv, const float* __restrict__ bv,
    __bf16* __restrict__ Qb, __bf16* __restrict__ Kb, __bf16* __restrict__ Vt)
{
  const int z = blockIdx.z;
  const float* W    = (z == 0) ? Wq : (z == 1) ? Wk : Wv;
  const float* bias = (z == 0) ? bq : (z == 1) ? bk : bv;

  const int t = threadIdx.x, lane = t & 31, wave = t >> 5;
  const int m0 = blockIdx.x * 64  + (wave >> 2) * 32;
  const int n0 = blockIdx.y * 128 + (wave & 3) * 32;
  const int lr = lane & 15;
  const int kb = (lane >> 4) << 3;

  v8f acc[2][2];
#pragma unroll
  for (int i = 0; i < 2; ++i)
#pragma unroll
    for (int j = 0; j < 2; ++j) acc[i][j] = (v8f)0.0f;

  for (int k0 = 0; k0 < DIM; k0 += 32) {
    const float* ax = x + (size_t)(m0 + lr) * DIM + (k0 + kb);
    const float* bw = W + (size_t)(n0 + lr) * DIM + (k0 + kb);
    v16bf a0 = frag_from_f32(ax);
    v16bf a1 = frag_from_f32(ax + 16 * DIM);
    v16bf b0 = frag_from_f32(bw);
    v16bf b1 = frag_from_f32(bw + 16 * DIM);
    acc[0][0] = wmma_bf16(a0, b0, acc[0][0]);
    acc[0][1] = wmma_bf16(a0, b1, acc[0][1]);
    acc[1][0] = wmma_bf16(a1, b0, acc[1][0]);
    acc[1][1] = wmma_bf16(a1, b1, acc[1][1]);
  }

  const int cn  = lane & 15;
  const int rhi = (lane >> 4) << 3;
#pragma unroll
  for (int mi = 0; mi < 2; ++mi)
#pragma unroll
    for (int ni = 0; ni < 2; ++ni) {
      const int ncol = n0 + ni * 16 + cn;
      const float bb = bias[ncol];
#pragma unroll
      for (int r = 0; r < 8; ++r) {
        const int mrow = m0 + mi * 16 + rhi + r;
        const __bf16 o = (__bf16)(acc[mi][ni][r] + bb);
        if (z == 2) {
          const int bI = mrow >> 11;
          const int lI = mrow & (SEQ - 1);
          Vt[((size_t)bI * DIM + ncol) * SEQ + lI] = o;
        } else if (z == 1) {
          Kb[(size_t)mrow * DIM + ncol] = o;
        } else {
          Qb[(size_t)mrow * DIM + ncol] = o;
        }
      }
    }
}

// ---------------------------------------------------------------------------
// Shared structure for the two attention GEMMs: block tile 64(M)x256(N),
// k-step 32, double-buffered LDS fed by global_load_async_to_lds_b128.
// 8 waves, each owning a 32x64 tile (2x4 fragments, 8 WMMA per k-step).
// Per tile each thread issues 5 async b128 copies (A:1, B:4).
// ---------------------------------------------------------------------------
#define ATTN_GEMM_BODY(ABASE, LDA_ELEMS, BBASE, LDB_ELEMS, NTILES, EPILOGUE)   \
  __shared__ __bf16 As[2][64 * 32];                                            \
  __shared__ __bf16 Bs[2][256 * 32];                                           \
  const int t = threadIdx.x, lane = t & 31, wave = t >> 5;                     \
  const int m0 = blockIdx.x * 64, n0 = blockIdx.y * 256;                       \
  const int wm = (wave & 1) * 32, wn = (wave >> 1) * 64;                       \
  const int lr = lane & 15, kb = (lane >> 4) << 3;                             \
  const int crow = t >> 2;              /* 0..63  */                           \
  const int ccol = (t & 3) * 8;         /* 0,8,16,24 elems */                  \
  const uint32_t aLds0 = lds_off_u32(&As[0][0]) + (uint32_t)t * 16u;           \
  const uint32_t bLds0 = lds_off_u32(&Bs[0][0]) + (uint32_t)t * 16u;           \
  v8f acc[2][4];                                                               \
  _Pragma("unroll") for (int i = 0; i < 2; ++i)                                \
      _Pragma("unroll") for (int j = 0; j < 4; ++j) acc[i][j] = (v8f)0.0f;     \
  /* issue tile 0 into buffer 0 */                                             \
  {                                                                            \
    const uint32_t k0 = 0;                                                     \
    async_b128(aLds0, (ABASE),                                                 \
               (uint32_t)(((m0 + crow) * (LDA_ELEMS) + k0 + ccol) * 2));       \
    _Pragma("unroll") for (int c = 0; c < 4; ++c)                              \
        async_b128(bLds0 + (uint32_t)c * 4096u, (BBASE),                       \
                   (uint32_t)(((n0 + 64 * c + crow) * (LDB_ELEMS) + k0 + ccol) \
                              * 2));                                           \
  }                                                                            \
  for (int kt = 0; kt < (NTILES); ++kt) {                                      \
    const int cur = kt & 1;                                                    \
    if (kt + 1 < (NTILES)) {                                                   \
      const uint32_t k0 = (uint32_t)(kt + 1) * 32u;                            \
      const uint32_t dst = (uint32_t)(cur ^ 1);                                \
      async_b128(aLds0 + dst * (64u * 32u * 2u), (ABASE),                      \
                 (uint32_t)(((m0 + crow) * (LDA_ELEMS) + k0 + ccol) * 2));     \
      _Pragma("unroll") for (int c = 0; c < 4; ++c)                            \
          async_b128(bLds0 + dst * (256u * 32u * 2u) + (uint32_t)c * 4096u,    \
                     (BBASE),                                                  \
                     (uint32_t)(((n0 + 64 * c + crow) * (LDB_ELEMS) + k0 +     \
                                 ccol) * 2));                                  \
      asm volatile("s_wait_asynccnt 0x5" ::: "memory");                        \
    } else {                                                                   \
      asm volatile("s_wait_asynccnt 0x0" ::: "memory");                        \
    }                                                                          \
    __syncthreads();                                                           \
    const __bf16* Ab = &As[cur][0];                                            \
    const __bf16* Bb = &Bs[cur][0];                                            \
    v16bf a0 = frag_from_bf16(Ab + (wm + lr) * 32 + kb);                       \
    v16bf a1 = frag_from_bf16(Ab + (wm + 16 + lr) * 32 + kb);                  \
    v16bf bfr[4];                                                              \
    _Pragma("unroll") for (int ni = 0; ni < 4; ++ni)                           \
        bfr[ni] = frag_from_bf16(Bb + (wn + ni * 16 + lr) * 32 + kb);          \
    _Pragma("unroll") for (int ni = 0; ni < 4; ++ni) {                         \
      acc[0][ni] = wmma_bf16(a0, bfr[ni], acc[0][ni]);                         \
      acc[1][ni] = wmma_bf16(a1, bfr[ni], acc[1][ni]);                         \
    }                                                                          \
    __syncthreads();                                                           \
  }                                                                            \
  const int cn = lane & 15, rhi = (lane >> 4) << 3;                            \
  _Pragma("unroll") for (int mi = 0; mi < 2; ++mi)                             \
      _Pragma("unroll") for (int ni = 0; ni < 4; ++ni)                         \
          _Pragma("unroll") for (int r = 0; r < 8; ++r) {                      \
    const int m = m0 + wm + mi * 16 + rhi + r;                                 \
    const int n = n0 + wn + ni * 16 + cn;                                      \
    EPILOGUE;                                                                  \
  }

// ---------------------------------------------------------------------------
// Kernel 2: S = Q @ K^T * (1/sqrt(D)) + mask. Per-batch, M=N=SEQ, K=DIM.
// ---------------------------------------------------------------------------
__global__ __launch_bounds__(256) void attn_scores(
    const __bf16* __restrict__ Qb, const __bf16* __restrict__ Kb,
    const float* __restrict__ mask, float* __restrict__ S)
{
  const int bI = blockIdx.z;
  const __bf16* Q  = Qb + (size_t)bI * SEQ * DIM;
  const __bf16* Km = Kb + (size_t)bI * SEQ * DIM;
  const float scale = 0.03125f;   // 1/sqrt(1024)
  ATTN_GEMM_BODY(Q, DIM, Km, DIM, DIM / 32, {
    const size_t off = ((size_t)bI * SEQ + m) * SEQ + n;
    S[off] = acc[mi][ni][r] * scale + mask[off];
  })
}

// ---------------------------------------------------------------------------
// Kernel 3: row softmax; P (bf16) written in place at the start of each f32
// logit row (all reads of the row precede the first barrier => safe).
// ---------------------------------------------------------------------------
__global__ __launch_bounds__(256) void softmax_rows(float* __restrict__ S)
{
  __shared__ float red[256];
  float* srow = S + (size_t)blockIdx.x * SEQ;
  __bf16* prow = (__bf16*)srow;   // row byte-stride stays SEQ*4
  const int t = threadIdx.x;

  float v[8];
#pragma unroll
  for (int i = 0; i < 8; ++i) v[i] = srow[t * 8 + i];

  float lmax = v[0];
#pragma unroll
  for (int i = 1; i < 8; ++i) lmax = fmaxf(lmax, v[i]);
  red[t] = lmax;
  __syncthreads();
  for (int s = 128; s > 0; s >>= 1) {
    if (t < s) red[t] = fmaxf(red[t], red[t + s]);
    __syncthreads();
  }
  const float rmax = red[0];
  __syncthreads();

  float lsum = 0.0f;
#pragma unroll
  for (int i = 0; i < 8; ++i) { v[i] = __expf(v[i] - rmax); lsum += v[i]; }
  red[t] = lsum;
  __syncthreads();
  for (int s = 128; s > 0; s >>= 1) {
    if (t < s) red[t] += red[t + s];
    __syncthreads();
  }
  const float inv = 1.0f / red[0];
#pragma unroll
  for (int i = 0; i < 8; ++i) prow[t * 8 + i] = (__bf16)(v[i] * inv);
}

// ---------------------------------------------------------------------------
// Kernel 4: O = P @ V. Per-batch, M=SEQ, N=DIM, K=SEQ. A = P bf16 rows aliased
// at f32-row starts (ld = 2*SEQ bf16 elems), B = Vt (K-contiguous).
// ---------------------------------------------------------------------------
__global__ __launch_bounds__(256) void attn_out(
    const float* __restrict__ Sp, const __bf16* __restrict__ Vt,
    float* __restrict__ out)
{
  const int bI = blockIdx.z;
  const __bf16* P  = (const __bf16*)(Sp + (size_t)bI * SEQ * SEQ);
  const __bf16* Vb = Vt + (size_t)bI * DIM * SEQ;
  ATTN_GEMM_BODY(P, 2 * SEQ, Vb, SEQ, SEQ / 32, {
    out[((size_t)bI * SEQ + m) * DIM + n] = acc[mi][ni][r];
  })
}

// ---------------------------------------------------------------------------
extern "C" void kernel_launch(void* const* d_in, const int* in_sizes, int n_in,
                              void* d_out, int out_size, void* d_ws, size_t ws_size,
                              hipStream_t stream) {
  (void)in_sizes; (void)n_in; (void)out_size; (void)ws_size;
  const float* x    = (const float*)d_in[0];
  const float* mask = (const float*)d_in[1];
  const float* Wq   = (const float*)d_in[2];
  const float* bq   = (const float*)d_in[3];
  const float* Wk   = (const float*)d_in[4];
  const float* bk   = (const float*)d_in[5];
  const float* Wv   = (const float*)d_in[6];
  const float* bv   = (const float*)d_in[7];
  float* out = (float*)d_out;

  // Workspace layout (~235 MB): Qb | Kb | Vt (bf16) | S (f32, P aliased in place)
  __bf16* Qb = (__bf16*)d_ws;
  __bf16* Kb = Qb + (size_t)BATCH * SEQ * DIM;
  __bf16* Vt = Kb + (size_t)BATCH * SEQ * DIM;
  float*  S  = (float*)(Vt + (size_t)BATCH * SEQ * DIM);

  dim3 blk(256);
  proj_qkv<<<dim3((BATCH * SEQ) / 64, DIM / 128, 3), blk, 0, stream>>>(
      x, Wq, bq, Wk, bk, Wv, bv, Qb, Kb, Vt);
  attn_scores<<<dim3(SEQ / 64, SEQ / 256, BATCH), blk, 0, stream>>>(Qb, Kb, mask, S);
  softmax_rows<<<dim3(BATCH * SEQ), blk, 0, stream>>>(S);
  attn_out<<<dim3(SEQ / 64, DIM / 256, BATCH), blk, 0, stream>>>(S, Vt, out);
}
